// PolicyHead_46566035423995
// MI455X (gfx1250) — compile-verified, matrix operand order checked by reading, not run on hardware
//
#include <hip/hip_runtime.h>

typedef __attribute__((ext_vector_type(16))) _Float16 v16h;
typedef __attribute__((ext_vector_type(8)))  float    v8f;

#define NBATCH 2048
#define HW     361        // 19*19
#define CINCH  256
#define CC     32
#define EPSB   1e-3f
#define MT     23         // ceil(361/16)
#define PITCH  392        // f16 pitch: 784 B = 196 DW; 196%64=4 -> conflict-free b128 stores
#define OUTROW 362

__global__ __launch_bounds__(256) void policy_head_fused(
    const float* __restrict__ x,       // [N,19,19,256]
    const float* __restrict__ w_p,     // [256,32]
    const float* __restrict__ w_g,     // [256,32]
    const float* __restrict__ beta_g,  // [32]
    const float* __restrict__ mean_g,  // [32]
    const float* __restrict__ var_g,   // [32]
    const float* __restrict__ w_bias,  // [64,32]
    const float* __restrict__ b_bias,  // [32]
    const float* __restrict__ beta_p,  // [32]
    const float* __restrict__ mean_p,  // [32]
    const float* __restrict__ var_p,   // [32]
    const float* __restrict__ w_out,   // [32,2]
    const float* __restrict__ w_pass,  // [64,2]
    const float* __restrict__ b_pass,  // [2]
    float* __restrict__ out)           // [2048,362] main ++ [2048,362] aux
{
    // channel-major conv outputs: rows 0..31 = p-channels, 32..63 = g-channels
    __shared__ _Float16 pgT[64 * PITCH];
    __shared__ float g_pooled[64];       // [mean(32), max(32)]
    __shared__ float bias_sh[CC];
    __shared__ float pass_sh[2];
    __shared__ float red_sum[256];
    __shared__ float red_max[256];

    const int tid   = threadIdx.x;
    const int lane  = tid & 31;
    const int wave  = tid >> 5;
    const int n     = blockIdx.x;
    const int khalf = lane >> 4;
    const int l16   = lane & 15;

    // ---------------- Phase 1: per-wave B fragments (registers, whole kernel) ----
    // weight matrix W[256][64] = [w_p | w_g]; wave owns N-tile (wave&3)
    const int ntile = wave & 3;
    const int mhalf = wave >> 2;
    const int c0    = ntile * 16;
    const float* wsrc = (ntile < 2) ? w_p : w_g;
    const int wcol = (ntile & 1) * 16 + l16;

    v16h bfrag[8];
    #pragma unroll
    for (int s = 0; s < 8; ++s) {
        #pragma unroll
        for (int r = 0; r < 8; ++r) {
            const int k = s * 32 + (r & 3) * 2 + (r >> 2) * 16 + khalf * 8;
            bfrag[s][2 * r]     = (_Float16)wsrc[(size_t)k       * CC + wcol];
            bfrag[s][2 * r + 1] = (_Float16)wsrc[(size_t)(k + 1) * CC + wcol];
        }
    }

    // ---------------- Phase 2: WMMA main loop (branch-free body) ----------------
    // Rows >= 361 are clamped to row 360: WMMA row m of D depends only on row m
    // of A, and D rows >= 361 are never read downstream, so no zero-fill needed.
    const size_t xbase = (size_t)n * HW * CINCH;
    for (int mt = mhalf; mt < MT; mt += 2) {
        const int mrow = mt * 16 + l16;
        const int mclamp = (mrow < HW) ? mrow : (HW - 1);
        const float* xrow = x + xbase + (size_t)mclamp * CINCH;
        v8f acc = {};
        #pragma unroll
        for (int s = 0; s < 8; ++s) {
            const int kb = s * 32 + khalf * 8;
            const float4 f0 = *(const float4*)(xrow + kb);
            const float4 f1 = *(const float4*)(xrow + kb + 4);
            const float4 f2 = *(const float4*)(xrow + kb + 16);
            const float4 f3 = *(const float4*)(xrow + kb + 20);
            v16h afrag;
            afrag[0]  = (_Float16)f0.x; afrag[1]  = (_Float16)f0.y;
            afrag[2]  = (_Float16)f0.z; afrag[3]  = (_Float16)f0.w;
            afrag[4]  = (_Float16)f1.x; afrag[5]  = (_Float16)f1.y;
            afrag[6]  = (_Float16)f1.z; afrag[7]  = (_Float16)f1.w;
            afrag[8]  = (_Float16)f2.x; afrag[9]  = (_Float16)f2.y;
            afrag[10] = (_Float16)f2.z; afrag[11] = (_Float16)f2.w;
            afrag[12] = (_Float16)f3.x; afrag[13] = (_Float16)f3.y;
            afrag[14] = (_Float16)f3.z; afrag[15] = (_Float16)f3.w;
            acc = __builtin_amdgcn_wmma_f32_16x16x32_f16(
                false, afrag, false, bfrag[s], (short)0, acc, false, false);
        }
        // D layout: VGPR r -> row mt*16 + 8*khalf + r (8 consecutive rows/lane),
        // channel c0 + l16.  Pack to 8 halves -> single 16B-aligned b128 store.
        union { uint4 u4; _Float16 h[8]; } pk;
        #pragma unroll
        for (int r = 0; r < 8; ++r) pk.h[r] = (_Float16)acc[r];
        const int mbase = mt * 16 + 8 * khalf;
        *(uint4*)&pgT[(c0 + l16) * PITCH + mbase] = pk.u4;
    }
    __syncthreads();

    // ---------------- Phase 3: BN+ReLU on g, mean/max pooling -------------------
    {
        const int c = tid >> 3;        // channel 0..31
        const int part = tid & 7;
        const float mg = mean_g[c];
        const float ig = rsqrtf(var_g[c] + EPSB);
        const float bg = beta_g[c];
        const _Float16* grow = &pgT[(32 + c) * PITCH];
        float s = 0.0f, mx = 0.0f;     // relu output >= 0, so 0 is a safe max seed
        for (int m = part; m < HW; m += 8) {
            float v = (float)grow[m];
            v = (v - mg) * ig + bg;
            v = fmaxf(v, 0.0f);
            s += v;
            mx = fmaxf(mx, v);
        }
        red_sum[tid] = s;
        red_max[tid] = mx;
    }
    __syncthreads();
    if (tid < 32) {
        float s = 0.0f;
        #pragma unroll
        for (int i = 0; i < 8; ++i) s += red_sum[tid * 8 + i];
        g_pooled[tid] = s * (1.0f / 361.0f);
    } else if (tid < 64) {
        const int c = tid - 32;
        float mx = 0.0f;
        #pragma unroll
        for (int i = 0; i < 8; ++i) mx = fmaxf(mx, red_max[c * 8 + i]);
        g_pooled[32 + c] = mx;
    }
    __syncthreads();

    // ---------------- Phase 4: pooled bias + pass logits ------------------------
    if (tid < 32) {
        float acc = b_bias[tid];
        #pragma unroll 8
        for (int j = 0; j < 64; ++j) acc += g_pooled[j] * w_bias[j * CC + tid];
        bias_sh[tid] = acc;
    } else if (tid < 34) {
        const int k = tid - 32;
        float acc = b_pass[k] - 3.0f;
        #pragma unroll 8
        for (int j = 0; j < 64; ++j) acc += g_pooled[j] * w_pass[j * 2 + k];
        pass_sh[k] = acc;
    }
    __syncthreads();

    // ---------------- Phase 5: BN+ReLU on p, project to 2 logits ----------------
    // lane == channel (32 channels, wave32): cross-lane reduce gives the logits
    const float bi = bias_sh[lane];
    const float mp = mean_p[lane];
    const float ip = rsqrtf(var_p[lane] + EPSB);
    const float bp = beta_p[lane];
    const float w0 = w_out[lane * 2 + 0];
    const float w1 = w_out[lane * 2 + 1];
    const _Float16* prow = &pgT[lane * PITCH];
    float* out_main = out;
    float* out_aux  = out + (size_t)NBATCH * OUTROW;

    for (int m = wave; m < HW; m += 8) {
        float v = (float)prow[m];
        v = (v + bi - mp) * ip + bp;
        v = fmaxf(v, 0.0f);
        float t0 = v * w0;
        float t1 = v * w1;
        #pragma unroll
        for (int off = 16; off > 0; off >>= 1) {
            t0 += __shfl_xor(t0, off, 32);
            t1 += __shfl_xor(t1, off, 32);
        }
        if (lane == 0) {
            out_main[(size_t)n * OUTROW + m] = t0;
            out_aux [(size_t)n * OUTROW + m] = t1;
        }
    }
    if (tid == 0) {
        out_main[(size_t)n * OUTROW + HW] = pass_sh[0];
        out_aux [(size_t)n * OUTROW + HW] = pass_sh[1];
    }
}

extern "C" void kernel_launch(void* const* d_in, const int* in_sizes, int n_in,
                              void* d_out, int out_size, void* d_ws, size_t ws_size,
                              hipStream_t stream) {
    (void)in_sizes; (void)n_in; (void)out_size; (void)d_ws; (void)ws_size;
    policy_head_fused<<<dim3(NBATCH), dim3(256), 0, stream>>>(
        (const float*)d_in[0],  (const float*)d_in[1],  (const float*)d_in[2],
        (const float*)d_in[3],  (const float*)d_in[4],  (const float*)d_in[5],
        (const float*)d_in[6],  (const float*)d_in[7],  (const float*)d_in[8],
        (const float*)d_in[9],  (const float*)d_in[10], (const float*)d_in[11],
        (const float*)d_in[12], (const float*)d_in[13],
        (float*)d_out);
}
